// Attention_79800492360199
// MI455X (gfx1250) — matrix operand order repaired and checked by reading, not compiled, not run on hardware
//
#include <hip/hip_runtime.h>
#include <hip/hip_bf16.h>
#include <math.h>

// ---------------------------------------------------------------------------
// Types / helpers
// ---------------------------------------------------------------------------
typedef __attribute__((ext_vector_type(16))) __bf16 v16bf;
typedef __attribute__((ext_vector_type(8)))  float  v8f;
typedef int v4i_ __attribute__((vector_size(16)));

union alignas(32) Frag16 {
    unsigned int u[8];
    v16bf        v;
};

__device__ __forceinline__ unsigned short f2bf(float f) {
    __bf16 h = (__bf16)f;
    return __builtin_bit_cast(unsigned short, h);
}

__device__ __forceinline__ v8f wmma_bf16(const Frag16& a, const Frag16& b, v8f c) {
    return __builtin_amdgcn_wmma_f32_16x16x32_bf16(
        /*neg_a=*/false, a.v, /*neg_b=*/false, b.v,
        /*c_mod=*/(short)0, c, /*reuse_a=*/false, /*reuse_b=*/false);
}

// 16-bit A-operand K offset for packed dword d (0..7), contiguous order:
// lane<16 holds K pairs {0,2,...,14}; lane>=16 the same +16.
__device__ __forceinline__ int a_koff(int d, int half) {
    return 2 * d + half * 16;
}

// ---------------------------------------------------------------------------
// gfx1250 async global->LDS copy path (ASYNCcnt-tracked), with a synchronous
// register-staged fallback if the builtin is not declared.
// Probe-learned signature: (v4i* global_src, v4i* lds_dst, imm offset, cpol).
// ---------------------------------------------------------------------------
#if defined(__has_builtin)
#  if __has_builtin(__builtin_amdgcn_global_load_async_to_lds_b128)
#    define HAVE_ASYNC_LDS 1
#  endif
#endif
#ifndef HAVE_ASYNC_LDS
#  define HAVE_ASYNC_LDS 0
#endif

__device__ __forceinline__ void cp16(const unsigned short* g, unsigned short* l) {
#if HAVE_ASYNC_LDS
    __builtin_amdgcn_global_load_async_to_lds_b128(
        (v4i_*)g, (v4i_*)l, 0, 0);
#else
    *reinterpret_cast<uint4*>(l) = *reinterpret_cast<const uint4*>(g);
#endif
}

__device__ __forceinline__ void cp_commit() {
#if HAVE_ASYNC_LDS
#  if __has_builtin(__builtin_amdgcn_s_wait_asynccnt)
    __builtin_amdgcn_s_wait_asynccnt(0);
#  else
    asm volatile("s_wait_asynccnt 0x0" ::: "memory");
#  endif
#endif
}

// ---------------------------------------------------------------------------
// f32 -> bf16 conversion
// ---------------------------------------------------------------------------
__global__ void conv_f32_bf16(const float* __restrict__ src,
                              unsigned short* __restrict__ dst, int n) {
    int i = blockIdx.x * blockDim.x + threadIdx.x;
    int stride = gridDim.x * blockDim.x;
    for (; i < n; i += stride) dst[i] = f2bf(src[i]);
}

// ---------------------------------------------------------------------------
// Tiled bf16 GEMM:  C(M,N) = A(M,K) * B(N,K)^T   (both row-major, bf16)
// Block: 256 threads = 8 waves; tile 128x64x64, double-buffered LDS.
// Wave = 2x2 WMMA frags (32x32), 8 WMMA per k-stage.
// MODE_QKV epilogue: Q scaled by (1/8)*log2(e) (exp2-space softmax),
//                    K to (b,h,s,dh); V transposed to (b,h,dh,s).
// MODE_OUT epilogue: f32 to outf.
// ---------------------------------------------------------------------------
constexpr int MODE_QKV = 0;
constexpr int MODE_OUT = 1;

template <int MODE>
__global__ __launch_bounds__(256, 2) void gemm_bf16_kernel(
    const unsigned short* __restrict__ A,
    const unsigned short* __restrict__ B,
    int M, int N, int K,
    unsigned short* __restrict__ qb,
    unsigned short* __restrict__ kb,
    unsigned short* __restrict__ vb,
    float* __restrict__ outf)
{
    __shared__ unsigned short As[2][128][72];  // 144B rows (16B aligned)
    __shared__ unsigned short Bs[2][64][72];

    const int m0   = blockIdx.y * 128;
    const int n0   = blockIdx.x * 64;
    const int tid  = threadIdx.x;
    const int wid  = tid >> 5;
    const int lane = tid & 31;
    const int half = lane >> 4;
    const int wm   = wid & 3;   // 4 waves along M (32 rows each)
    const int wn   = wid >> 2;  // 2 waves along N (32 cols each)

    const v8f vzero = {0.f, 0.f, 0.f, 0.f, 0.f, 0.f, 0.f, 0.f};
    v8f acc[2][2];
    acc[0][0] = vzero; acc[0][1] = vzero; acc[1][0] = vzero; acc[1][1] = vzero;

    const int lr = tid >> 3;        // 0..31
    const int lc = (tid & 7) * 8;   // 0..56 (bf16 elems; 16B chunk)

    auto issue = [&](int kt, int buf) {
        const int k0 = kt * 64;
#pragma unroll
        for (int p = 0; p < 4; ++p) {
            int r = lr + p * 32;
            cp16(&A[(size_t)(m0 + r) * K + k0 + lc], &As[buf][r][lc]);
        }
#pragma unroll
        for (int p = 0; p < 2; ++p) {
            int r = lr + p * 32;
            cp16(&B[(size_t)(n0 + r) * K + k0 + lc], &Bs[buf][r][lc]);
        }
    };

    const int nk = K >> 6;
    int cur = 0;
    issue(0, 0);

    for (int kt = 0; kt < nk; ++kt) {
        cp_commit();
        __syncthreads();
        if (kt + 1 < nk) issue(kt + 1, cur ^ 1);

        Frag16 af[2][2], bf[2][2];
#pragma unroll
        for (int kc = 0; kc < 2; ++kc) {
#pragma unroll
            for (int sm = 0; sm < 2; ++sm) {
                int m = wm * 32 + sm * 16 + (lane & 15);
#pragma unroll
                for (int d = 0; d < 8; ++d)
                    af[kc][sm].u[d] = *reinterpret_cast<const unsigned int*>(
                        &As[cur][m][kc * 32 + a_koff(d, half)]);
            }
#pragma unroll
            for (int sn = 0; sn < 2; ++sn) {
                int n  = wn * 32 + sn * 16 + (lane & 15);
                int kk = kc * 32 + half * 16;
                *reinterpret_cast<uint4*>(&bf[kc][sn].u[0]) =
                    *reinterpret_cast<const uint4*>(&Bs[cur][n][kk]);
                *reinterpret_cast<uint4*>(&bf[kc][sn].u[4]) =
                    *reinterpret_cast<const uint4*>(&Bs[cur][n][kk + 8]);
            }
        }
#pragma unroll
        for (int kc = 0; kc < 2; ++kc)
#pragma unroll
            for (int sm = 0; sm < 2; ++sm)
#pragma unroll
                for (int sn = 0; sn < 2; ++sn)
                    acc[sm][sn] = wmma_bf16(af[kc][sm], bf[kc][sn], acc[sm][sn]);
        cur ^= 1;
    }

    // Epilogue
#pragma unroll
    for (int sm = 0; sm < 2; ++sm) {
#pragma unroll
        for (int sn = 0; sn < 2; ++sn) {
            v8f c = acc[sm][sn];
            int gn = n0 + wn * 32 + sn * 16 + (lane & 15);
#pragma unroll
            for (int r = 0; r < 8; ++r) {
                int gm = m0 + wm * 32 + sm * 16 + half * 8 + r;
                float val = c[r];
                if (MODE == MODE_OUT) {
                    outf[(size_t)gm * N + gn] = val;
                } else {
                    int part = gn >> 10;       // 0=q 1=k 2=v
                    int e    = gn & 1023;
                    int h    = e >> 6;
                    int dh   = e & 63;
                    int b    = gm >> 11;       // S = 2048
                    int s    = gm & 2047;
                    if (part == 2) {
                        // V stored transposed: (b, h, dh, s)
                        vb[(((size_t)b * 16 + h) * 64 + dh) * 2048 + s] =
                            f2bf(val);
                    } else {
                        // Fold 1/sqrt(64) and log2(e) (exp2 softmax) into Q.
                        if (part == 0) val *= 0.125f * 1.44269504088896340736f;
                        unsigned short* dst = (part == 0) ? qb : kb;
                        dst[(((size_t)b * 16 + h) * 2048 + s) * 64 + dh] =
                            f2bf(val);
                    }
                }
            }
        }
    }
}

// ---------------------------------------------------------------------------
// Causal flash attention (exp2-space online softmax).
// Block: 128 threads = 4 waves, 64 query rows (16 per wave).
// grid.x = B*H*(S/64) = 2048.  Q/K bf16 (b,h,s,64); V bf16 transposed
// (b,h,64,s); Q pre-scaled by log2(e)/8.  Double-buffered K/V tiles.
// Diagonal tile peeled so the steady-state loop carries no mask code.
// Output: bf16 (b, s, 1024) with d = h*64 + dh.
// ---------------------------------------------------------------------------
__global__ __launch_bounds__(128, 2) void attn_kernel(
    const unsigned short* __restrict__ Q,
    const unsigned short* __restrict__ Kb,
    const unsigned short* __restrict__ Vtb,
    unsigned short* __restrict__ Ob)
{
    const int S = 2048, DH = 64;
    const int blk = blockIdx.x;
    const int qt  = blk & 31;     // query tile index (S/64 = 32)
    const int bh  = blk >> 5;     // batch*head, 0..63
    const int b   = bh >> 4;
    const int h   = bh & 15;

    const unsigned short* q  = Q   + (size_t)bh * S * DH;
    const unsigned short* k  = Kb  + (size_t)bh * S * DH;
    const unsigned short* vt = Vtb + (size_t)bh * DH * S;  // (dh, s)

    const int tid  = threadIdx.x;
    const int wid  = tid >> 5;
    const int lane = tid & 31;
    const int half = lane >> 4;
    const int qrow0 = qt * 64 + wid * 16;   // first query row of this wave

    __shared__ unsigned short Ks[2][64][72];    // [s'][d]   144B rows
    __shared__ unsigned short Vts[2][64][72];   // [d][s']   (V pre-transposed)
    __shared__ unsigned short Ps[4][16][72];    // per-wave P staging

    // Q fragments (A operand), 2 K-chunks over DH=64, loaded once.
    Frag16 aq[2];
    {
        int m = qrow0 + (lane & 15);
        const unsigned short* qrow = q + (size_t)m * DH;
#pragma unroll
        for (int kc = 0; kc < 2; ++kc)
#pragma unroll
            for (int d = 0; d < 8; ++d) {
                int kk = kc * 32 + a_koff(d, half);
                aq[kc].u[d] =
                    *reinterpret_cast<const unsigned int*>(&qrow[kk]);
            }
    }

    const v8f vzero = {0.f, 0.f, 0.f, 0.f, 0.f, 0.f, 0.f, 0.f};
    float m_run[8], l_run[8];
    v8f o[4];
#pragma unroll
    for (int r = 0; r < 8; ++r) { m_run[r] = -INFINITY; l_run[r] = 0.f; }
#pragma unroll
    for (int nb = 0; nb < 4; ++nb) o[nb] = vzero;

    auto issue = [&](int kt, int buf) {
        int r = tid >> 3;            // 0..15
        int c = (tid & 7) * 8;       // 0..56
#pragma unroll
        for (int p = 0; p < 4; ++p) {
            int row = r + p * 16;
            cp16(&k[(size_t)(kt * 64 + row) * DH + c], &Ks[buf][row][c]);
            cp16(&vt[(size_t)row * S + kt * 64 + c], &Vts[buf][row][c]);
        }
    };

    int cur = 0;

    auto step = [&](int kt, bool diag) {
        cp_commit();
        __syncthreads();
        if (!diag) issue(kt + 1, cur ^ 1);

        // S = Q * K^T  (16 x 64 per wave); Q carries log2(e)/8 scale.
        v8f sacc[4];
#pragma unroll
        for (int nb = 0; nb < 4; ++nb) {
            sacc[nb] = vzero;
#pragma unroll
            for (int kc = 0; kc < 2; ++kc) {
                Frag16 bk;
                int n  = nb * 16 + (lane & 15);
                int kk = kc * 32 + half * 16;
                *reinterpret_cast<uint4*>(&bk.u[0]) =
                    *reinterpret_cast<const uint4*>(&Ks[cur][n][kk]);
                *reinterpret_cast<uint4*>(&bk.u[4]) =
                    *reinterpret_cast<const uint4*>(&Ks[cur][n][kk + 8]);
                sacc[nb] = wmma_bf16(aq[kc], bk, sacc[nb]);
            }
        }

        // Causal mask (diagonal tile only) + online softmax in exp2 space.
        float pmax[8];
#pragma unroll
        for (int r = 0; r < 8; ++r) pmax[r] = m_run[r];
        float logits[4][8];
#pragma unroll
        for (int nb = 0; nb < 4; ++nb) {
#pragma unroll
            for (int r = 0; r < 8; ++r) {
                float t = sacc[nb][r];
                if (diag) {
                    int col = kt * 64 + nb * 16 + (lane & 15);
                    int row = qrow0 + half * 8 + r;
                    if (col > row) t = -INFINITY;
                }
                logits[nb][r] = t;
                pmax[r] = fmaxf(pmax[r], t);
            }
        }
        // Row max across the 16-lane column group.
#pragma unroll
        for (int mk = 1; mk < 16; mk <<= 1)
#pragma unroll
            for (int r = 0; r < 8; ++r)
                pmax[r] = fmaxf(pmax[r], __shfl_xor(pmax[r], mk, 32));

        float alpha[8], psum[8];
#pragma unroll
        for (int r = 0; r < 8; ++r) {
            alpha[r] = exp2f(m_run[r] - pmax[r]);
            psum[r]  = 0.f;
        }
        float p[4][8];
#pragma unroll
        for (int nb = 0; nb < 4; ++nb)
#pragma unroll
            for (int r = 0; r < 8; ++r) {
                float e = exp2f(logits[nb][r] - pmax[r]);
                p[nb][r] = e;
                psum[r] += e;
            }
#pragma unroll
        for (int mk = 1; mk < 16; mk <<= 1)
#pragma unroll
            for (int r = 0; r < 8; ++r)
                psum[r] += __shfl_xor(psum[r], mk, 32);
#pragma unroll
        for (int r = 0; r < 8; ++r) {
            l_run[r] = l_run[r] * alpha[r] + psum[r];
            m_run[r] = pmax[r];
        }
#pragma unroll
        for (int nb = 0; nb < 4; ++nb)
#pragma unroll
            for (int r = 0; r < 8; ++r) o[nb][r] *= alpha[r];

        // Stage P (C/D layout) -> LDS -> reload in A layout. Same-wave DS ops
        // are in-order; the asm barrier stops compiler reordering.
#pragma unroll
        for (int nb = 0; nb < 4; ++nb)
#pragma unroll
            for (int r = 0; r < 8; ++r)
                Ps[wid][half * 8 + r][nb * 16 + (lane & 15)] = f2bf(p[nb][r]);
        asm volatile("" ::: "memory");

        Frag16 ap[2];
        {
            int m = lane & 15;
#pragma unroll
            for (int kc = 0; kc < 2; ++kc)
#pragma unroll
                for (int d = 0; d < 8; ++d) {
                    int kk = kc * 32 + a_koff(d, half);
                    ap[kc].u[d] =
                        *reinterpret_cast<const unsigned int*>(&Ps[wid][m][kk]);
                }
        }

        // O += P * V   (B operand from pre-transposed V: contiguous along s')
#pragma unroll
        for (int nb = 0; nb < 4; ++nb) {
#pragma unroll
            for (int kc = 0; kc < 2; ++kc) {
                Frag16 bv;
                int n  = nb * 16 + (lane & 15);
                int kk = kc * 32 + half * 16;
                *reinterpret_cast<uint4*>(&bv.u[0]) =
                    *reinterpret_cast<const uint4*>(&Vts[cur][n][kk]);
                *reinterpret_cast<uint4*>(&bv.u[4]) =
                    *reinterpret_cast<const uint4*>(&Vts[cur][n][kk + 8]);
                o[nb] = wmma_bf16(ap[kc], bv, o[nb]);
            }
        }
        cur ^= 1;
    };

    issue(0, 0);
    for (int kt = 0; kt < qt; ++kt) step(kt, false);  // mask-free steady state
    step(qt, true);                                   // peeled diagonal tile

    // Normalize and write out: (b, s, h*64+dh) as bf16.
#pragma unroll
    for (int nb = 0; nb < 4; ++nb) {
        int dh = nb * 16 + (lane & 15);
#pragma unroll
        for (int r = 0; r < 8; ++r) {
            int s = qrow0 + half * 8 + r;
            float val = o[nb][r] / l_run[r];
            Ob[((size_t)b * 2048 + s) * 1024 + h * 64 + dh] = f2bf(val);
        }
    }
}

// ---------------------------------------------------------------------------
// Launch
// ---------------------------------------------------------------------------
extern "C" void kernel_launch(void* const* d_in, const int* in_sizes, int n_in,
                              void* d_out, int out_size, void* d_ws, size_t ws_size,
                              hipStream_t stream) {
    const float* x    = (const float*)d_in[0];   // (4, 2048, 1024)
    const float* wqkv = (const float*)d_in[1];   // (3072, 1024)
    const float* wo   = (const float*)d_in[2];   // (1024, 1024)
    float* out = (float*)d_out;                  // (4, 2048, 1024)

    char* ws = (char*)d_ws;
    // Workspace layout (bytes):
    unsigned short* xb    = (unsigned short*)(ws + 0);          // 16 MB
    unsigned short* wqkvb = (unsigned short*)(ws + 16777216);   // 6 MB
    unsigned short* wob   = (unsigned short*)(ws + 23068672);   // 2 MB
    unsigned short* qb    = (unsigned short*)(ws + 25165824);   // 16 MB
    unsigned short* kb    = (unsigned short*)(ws + 41943040);   // 16 MB
    unsigned short* vtb   = (unsigned short*)(ws + 58720256);   // 16 MB (V^T)
    unsigned short* ab    = (unsigned short*)(ws + 75497472);   // 16 MB

    const int M = 4 * 2048;     // 8192 rows
    const int K = 1024;

    conv_f32_bf16<<<1024, 256, 0, stream>>>(x, xb, M * K);
    conv_f32_bf16<<<1024, 256, 0, stream>>>(wqkv, wqkvb, 3 * 1024 * 1024);
    conv_f32_bf16<<<512, 256, 0, stream>>>(wo, wob, 1024 * 1024);

    // QKV projection: (8192 x 1024) * (3072 x 1024)^T, scatter into Q/K/V^T.
    gemm_bf16_kernel<MODE_QKV><<<dim3(3072 / 64, 8192 / 128), 256, 0, stream>>>(
        xb, wqkvb, M, 3072, K, qb, kb, vtb, nullptr);

    // Causal flash attention over 64 batch-heads x 32 query tiles.
    attn_kernel<<<2048, 128, 0, stream>>>(qb, kb, vtb, ab);

    // Output projection: (8192 x 1024) * (1024 x 1024)^T -> f32 out.
    gemm_bf16_kernel<MODE_OUT><<<dim3(1024 / 64, 8192 / 128), 256, 0, stream>>>(
        ab, wob, M, 1024, K, nullptr, nullptr, nullptr, out);
}
